// Baseline_35940286333526
// MI455X (gfx1250) — matrix operand order repaired and checked by reading, not compiled, not run on hardware
//
#include <hip/hip_runtime.h>
#include <cstdint>
#include <cstddef>

// ---------------------------------------------------------------------------
// tanh-RNN for MI455X (gfx1250, wave32, WMMA bf16 16x16x32, f32 accumulate)
//
//   phase 0: repack U_w (512x512) and W_w (256x512) f32 -> bf16 in the exact
//            WMMA B-fragment lane order (32 contiguous bytes per lane).
//   phase 1: wx = x @ W + W_b as bf16, stored in C-fragment order
//            [t][g][nt][lane][8]  (g = batch/16, nt = hidden/16). Full-GPU
//            parallel WMMA GEMM, native packed f32->bf16 converts.
//   phase 2: 4 persistent workgroups (one per 16-row batch group), 16 waves
//            each. h (16x512 bf16) double-buffered in LDS; per step each wave
//            owns 2 N-tiles: 32 v_wmma_f32_16x16x32_bf16 over K=512, fast
//            tanh, scatter h_next to LDS, one workgroup barrier. U streamed
//            from L2 every step via coalesced b128 loads (the per-step
//            bandwidth floor). Tail: h@V_w + V_b, sigmoid.
//
// d_ws layout (needs ~135 MB):
//   [0, 128MB)          wx packed bf16  (64*2048*512 elems)
//   [128MB, +512KB)     U packed bf16   (16*32*32*16 elems)
//   [+512KB, +256KB)    W packed bf16   ( 8*32*32*16 elems)
// ---------------------------------------------------------------------------

typedef __bf16        bf16x16 __attribute__((ext_vector_type(16)));
typedef __bf16        bf16x8  __attribute__((ext_vector_type(8)));
typedef float         f32x8   __attribute__((ext_vector_type(8)));
typedef unsigned int  u32x4   __attribute__((ext_vector_type(4)));

#define RNN_INPUT   256
#define RNN_HIDDEN  512
#define RNN_BATCH   64
#define RNN_SEQ     2048

// tanh via the hardware transcendental unit; exact at +/-inf saturation.
__device__ __forceinline__ float fast_tanh(float x) {
    float e = __expf(2.0f * x);          // v_exp_f32
    return 1.0f - 2.0f / (e + 1.0f);     // e=inf -> 1, e=0 -> -1
}

// ---------------------------------------------------------------------------
// Phase 0: repack U_w and W_w into WMMA B-fragment order.
// B-fragment (16x16x32 bf16): lane l holds column n = nt*16 + l%16,
// K = kk*32 + 8*(l/16) + e + (e&8), e = 0..15  -> 32 contiguous bytes per lane.
// ---------------------------------------------------------------------------
__global__ void repack_kernel(const float* __restrict__ U,
                              const float* __restrict__ W,
                              __bf16* __restrict__ Upk,
                              __bf16* __restrict__ Wpk) {
    unsigned idx = blockIdx.x * 256u + threadIdx.x;
    if (idx < 16u * 32u * 32u * 16u) {                       // 262144 U elems
        unsigned e    = idx & 15u;
        unsigned lane = (idx >> 4) & 31u;
        unsigned nt   = (idx >> 9) & 31u;
        unsigned kk   = idx >> 14;
        unsigned k    = kk * 32u + ((lane >> 4) << 3) + e + (e & 8u);
        unsigned n    = nt * 16u + (lane & 15u);
        Upk[idx] = (__bf16)U[k * (unsigned)RNN_HIDDEN + n];
    } else {
        unsigned j = idx - 262144u;
        if (j < 8u * 32u * 32u * 16u) {                      // 131072 W elems
            unsigned e    = j & 15u;
            unsigned lane = (j >> 4) & 31u;
            unsigned nt   = (j >> 9) & 31u;
            unsigned kk   = j >> 14;
            unsigned k    = kk * 32u + ((lane >> 4) << 3) + e + (e & 8u);   // k < 256
            unsigned n    = nt * 16u + (lane & 15u);
            Wpk[j] = (__bf16)W[k * (unsigned)RNN_HIDDEN + n];
        }
    }
}

// ---------------------------------------------------------------------------
// Phase 1: wx = x @ W + W_b, bf16, C-fragment packed [t][g][nt][lane][8].
// One 16x16 output tile per wave; tile id = (t, g, nt).
// A-fragment: lane l -> row M = l%16, K = kk*32 + 8*(l/16) + {0..7, 16..23}.
// ---------------------------------------------------------------------------
__global__ void __launch_bounds__(256)
wx_kernel(const float* __restrict__ x,
          const float* __restrict__ Wb,
          const __bf16* __restrict__ Wpk,
          __bf16* __restrict__ wxpk) {
    const unsigned lane = threadIdx.x & 31u;
    const unsigned wave = threadIdx.x >> 5;
    const unsigned id   = blockIdx.x * 8u + wave;       // 262144 tiles total
    const unsigned nt   = id & 31u;
    const unsigned g    = (id >> 5) & 3u;
    const unsigned t    = id >> 7;
    const unsigned col  = lane & 15u;
    const unsigned half = lane >> 4;

    const unsigned b = g * 16u + col;                   // batch row of this lane
    const float* xrow = x + ((size_t)b * RNN_SEQ + t) * RNN_INPUT;

    f32x8 acc = {0.f, 0.f, 0.f, 0.f, 0.f, 0.f, 0.f, 0.f};
#pragma unroll
    for (unsigned kk = 0; kk < RNN_INPUT / 32u; ++kk) {
        const float* p0 = xrow + kk * 32u + half * 8u;
        f32x8 x0 = *(const f32x8*)p0;                   // K = kk*32+8h+0..7
        f32x8 x1 = *(const f32x8*)(p0 + 16);            // K = kk*32+8h+16..23
        bf16x8 a0 = __builtin_convertvector(x0, bf16x8);   // v_cvt_pk_bf16_f32
        bf16x8 a1 = __builtin_convertvector(x1, bf16x8);
        bf16x16 a = __builtin_shufflevector(a0, a1, 0, 1, 2, 3, 4, 5, 6, 7,
                                            8, 9, 10, 11, 12, 13, 14, 15);
        const bf16x16 bf = *(const bf16x16*)(Wpk + (((kk * 32u + nt) * 32u + lane) << 4));
        acc = __builtin_amdgcn_wmma_f32_16x16x32_bf16(false, a, false, bf,
                                                      (short)0, acc, false, false);
    }
    const float bias = Wb[nt * 16u + col];
#pragma unroll
    for (int v = 0; v < 8; ++v) acc[v] += bias;
    bf16x8 ov = __builtin_convertvector(acc, bf16x8);
    *(bf16x8*)(wxpk + ((((size_t)t * 4u + g) * 32u + nt) * 256u + lane * 8u)) = ov;
}

// ---------------------------------------------------------------------------
// Phase 2: persistent scan. 4 blocks x 512 threads (16 waves). Block g owns
// batch rows [16g, 16g+16). h double-buffered in LDS with 520-elem row
// stride (1040 B = 260 dwords -> <=2-way bank conflicts on b128 row reads).
// ---------------------------------------------------------------------------
__global__ void __launch_bounds__(512)
scan_kernel(const __bf16* __restrict__ wxpk,
            const __bf16* __restrict__ Upk,
            const float* __restrict__ Ub,
            const float* __restrict__ Vw,
            const float* __restrict__ Vb,
            float* __restrict__ out) {
    constexpr unsigned RS = 520;                       // padded row stride
    __shared__ __align__(16) __bf16 hbuf[2][16 * RS];

    const unsigned tid  = threadIdx.x;
    const unsigned lane = tid & 31u;
    const unsigned wave = tid >> 5;
    const unsigned g    = blockIdx.x;
    const unsigned col  = lane & 15u;
    const unsigned half = lane >> 4;
    const unsigned nt0  = wave * 2u;
    const unsigned nt1  = nt0 + 1u;

    for (unsigned i = tid; i < 16u * RS; i += 512u) hbuf[0][i] = (__bf16)0.0f;
    __syncthreads();

    const float ub0 = Ub[nt0 * 16u + col];
    const float ub1 = Ub[nt1 * 16u + col];

    for (unsigned t = 0; t < RNN_SEQ; ++t) {
        const unsigned cur = t & 1u, nxt = cur ^ 1u;
        const size_t   wb  = ((size_t)t * 4u + g) * 32u;

        // acc = wx_t (bf16 C-fragment) + U_b
        f32x8 acc0, acc1;
        {
            bf16x8 w0 = *(const bf16x8*)(wxpk + (wb + nt0) * 256u + lane * 8u);
            bf16x8 w1 = *(const bf16x8*)(wxpk + (wb + nt1) * 256u + lane * 8u);
#pragma unroll
            for (int v = 0; v < 8; ++v) {
                acc0[v] = (float)w0[v] + ub0;
                acc1[v] = (float)w1[v] + ub1;
            }
        }

        // acc += h_cur @ U  (K = 512 -> 16 k-chunks of 32)
#pragma unroll 4
        for (unsigned kk = 0; kk < 16u; ++kk) {
            const unsigned ha = col * RS + kk * 32u + half * 8u;
            bf16x8 alo = *(const bf16x8*)&hbuf[cur][ha];        // ds_load_b128
            bf16x8 ahi = *(const bf16x8*)&hbuf[cur][ha + 16u];  // ds_load_b128
            bf16x16 a = __builtin_shufflevector(alo, ahi, 0, 1, 2, 3, 4, 5, 6, 7,
                                                8, 9, 10, 11, 12, 13, 14, 15);
            const bf16x16 b0 = *(const bf16x16*)(Upk + (((kk * 32u + nt0) * 32u + lane) << 4));
            const bf16x16 b1 = *(const bf16x16*)(Upk + (((kk * 32u + nt1) * 32u + lane) << 4));
            acc0 = __builtin_amdgcn_wmma_f32_16x16x32_bf16(false, a, false, b0,
                                                           (short)0, acc0, false, false);
            acc1 = __builtin_amdgcn_wmma_f32_16x16x32_bf16(false, a, false, b1,
                                                           (short)0, acc1, false, false);
        }

        // h_next = tanh(acc); scatter into next LDS buffer (C-fragment layout)
#pragma unroll
        for (int v = 0; v < 8; ++v) {
            const unsigned row = (unsigned)v + 8u * half;
            hbuf[nxt][row * RS + nt0 * 16u + col] = (__bf16)fast_tanh(acc0[v]);
            hbuf[nxt][row * RS + nt1 * 16u + col] = (__bf16)fast_tanh(acc1[v]);
        }
        __syncthreads();   // one barrier/step; double buffer makes this safe
    }

    // Output: o = sigmoid(h_T @ V_w + V_b). Final h is in hbuf[0] (SEQ even).
    if (tid < 32u) {
        const unsigned r = tid >> 1, c = tid & 1u;
        float s = Vb[c];
        for (unsigned k = 0; k < RNN_HIDDEN; ++k)
            s += (float)hbuf[0][r * RS + k] * Vw[k * 2u + c];
        out[(g * 16u + r) * 2u + c] = 1.f / (1.f + __expf(-s));
    }
}

// ---------------------------------------------------------------------------
extern "C" void kernel_launch(void* const* d_in, const int* in_sizes, int n_in,
                              void* d_out, int out_size, void* d_ws, size_t ws_size,
                              hipStream_t stream) {
    const float* x  = (const float*)d_in[0];   // [64,2048,256]
    const float* Ww = (const float*)d_in[1];   // [256,512]
    const float* Wb = (const float*)d_in[2];   // [512]
    const float* Uw = (const float*)d_in[3];   // [512,512]
    const float* Ub = (const float*)d_in[4];   // [512]
    const float* Vw = (const float*)d_in[5];   // [512,2]
    const float* Vb = (const float*)d_in[6];   // [2]
    float* out = (float*)d_out;                // [64,2] f32

    const size_t WX_BYTES = (size_t)RNN_BATCH * RNN_SEQ * RNN_HIDDEN * 2;  // 128 MB
    __bf16* wxpk = (__bf16*)d_ws;
    __bf16* Upk  = (__bf16*)((char*)d_ws + WX_BYTES);
    __bf16* Wpk  = (__bf16*)((char*)d_ws + WX_BYTES + 524288);

    // phase 0: weight repack (262144 + 131072 elements, 256 thr/block)
    repack_kernel<<<1536, 256, 0, stream>>>(Uw, Ww, Upk, Wpk);
    // phase 1: wx GEMM, 262144 tiles, 8 waves (tiles) per 256-thread block
    wx_kernel<<<32768, 256, 0, stream>>>(x, Wb, Wpk, wxpk);
    // phase 2: 4 independent persistent batch-group scans
    scan_kernel<<<4, 512, 0, stream>>>(wxpk, Upk, Ub, Vw, Vb, out);
}
